// GraphVAE_76347338654297
// MI455X (gfx1250) — matrix operand order repaired and checked by reading, not compiled, not run on hardware
//
#include <hip/hip_runtime.h>
#include <hip/hip_bf16.h>

// GraphVAE forward for MI455X (gfx1250, wave32, WMMA).
// Roofline: 3 streaming passes over adj (3x256MB) + 256MB adj_recon write
// ~= 1GB -> ~45us at 23.3 TB/s. K=8192 matmuls need ~1 PFLOP/s to remain
// BW-bound => must use v_wmma_f32_16x16x32_bf16 (f32 accumulate, bf16
// multiplicands converted on the fly from the f32 adj stream).

#define NN   8192
#define FIN  64
#define HH   128
#define ZZ   32

typedef __attribute__((ext_vector_type(16))) __bf16 v16bf;
typedef __attribute__((ext_vector_type(8)))  __bf16 v8bf;
typedef __attribute__((ext_vector_type(8)))  float  v8f;

// ---------------------------------------------------------------------------
// Pass 0: deg = rowsum(adj) + 1 (self loop); dinv = deg^-1/2  (deg >= 1 here)
// ---------------------------------------------------------------------------
__global__ __launch_bounds__(256) void k_deg(const float* __restrict__ adj,
                                             float* __restrict__ dinv) {
  __shared__ float red[256];
  const int row = blockIdx.x;
  const float4* ap = (const float4*)(adj + (size_t)row * NN);
  float s = 0.0f;
#pragma unroll
  for (int i = 0; i < 8; ++i) {
    float4 v = ap[threadIdx.x + i * 256];
    s += v.x + v.y + v.z + v.w;
  }
  red[threadIdx.x] = s;
  __syncthreads();
  for (int off = 128; off > 0; off >>= 1) {
    if (threadIdx.x < off) red[threadIdx.x] += red[threadIdx.x + off];
    __syncthreads();
  }
  if (threadIdx.x == 0) dinv[row] = rsqrtf(1.0f + red[0]);
}

// ---------------------------------------------------------------------------
// xsT[c][k] = bf16(dinv[k] * x[k][c])   (transposed => B-frag loads contiguous)
// ---------------------------------------------------------------------------
__global__ __launch_bounds__(256) void k_prep_x(const float* __restrict__ x,
                                                const float* __restrict__ dinv,
                                                __bf16* __restrict__ xsT) {
  const int idx = blockIdx.x * 256 + threadIdx.x;  // FIN * NN
  const int c = idx >> 13;                         // /8192
  const int k = idx & (NN - 1);
  xsT[(size_t)c * NN + k] = (__bf16)(dinv[k] * x[(size_t)k * FIN + c]);
}

// ---------------------------------------------------------------------------
// Big streaming GEMM: Y[row][col] = dinv[row] * sum_k bf16(adj[row][k]+I) * BT[col][k]
// Block = 8 waves, each wave owns 16 rows x FCOLS cols. K-loop step 32.
// A-frag per ISA 16-bit layout: lanes 0-15 K={0..7,16..23}, lanes 16-31
// K={8..15,24..31} (khalf = lane>>4 adds +8). B-frag: lane halves hold
// the same K sets as A for column (lane&15); BT transposed => contiguous loads.
// ---------------------------------------------------------------------------
template <int FCOLS>
__global__ __launch_bounds__(256) void k_gemm_big(const float* __restrict__ adj,
                                                  const __bf16* __restrict__ BT,
                                                  const float* __restrict__ dinv,
                                                  float* __restrict__ Y) {
  constexpr int G = FCOLS / 16;
  const int lane  = threadIdx.x & 31;
  const int wave  = threadIdx.x >> 5;
  const int m     = lane & 15;
  const int khalf = lane >> 4;
  const int row   = blockIdx.x * 128 + wave * 16 + m;

  const v8f vzero = {0.f, 0.f, 0.f, 0.f, 0.f, 0.f, 0.f, 0.f};
  v8f acc[G];
#pragma unroll
  for (int g = 0; g < G; ++g) acc[g] = vzero;

  const float* arow = adj + (size_t)row * NN;

#pragma unroll 1
  for (int kc = 0; kc < NN; kc += 32) {
    const float* ap = arow + kc + khalf * 8;
    __builtin_prefetch(ap + 64, 0, 1);  // K chunk two steps ahead -> global_prefetch_b8
    float4 a0 = *(const float4*)(ap);
    float4 a1 = *(const float4*)(ap + 4);
    float4 a2 = *(const float4*)(ap + 16);
    float4 a3 = *(const float4*)(ap + 20);
    float av[16] = {a0.x, a0.y, a0.z, a0.w, a1.x, a1.y, a1.z, a1.w,
                    a2.x, a2.y, a2.z, a2.w, a3.x, a3.y, a3.z, a3.w};
    if ((unsigned)(row - kc) < 32u) {  // diagonal (+I) fix-up, rare branch
      const int kb = kc + khalf * 8;
#pragma unroll
      for (int j = 0; j < 8; ++j) {
        if (row == kb + j)      av[j]     += 1.0f;
        if (row == kb + 16 + j) av[8 + j] += 1.0f;
      }
    }
    v16bf af;
#pragma unroll
    for (int j = 0; j < 16; ++j) af[j] = (__bf16)av[j];

    const int kb2 = kc + khalf * 8;
#pragma unroll
    for (int g = 0; g < G; ++g) {
      const __bf16* bp = BT + (size_t)(g * 16 + m) * NN + kb2;
      v8bf blo = *(const v8bf*)(bp);
      v8bf bhi = *(const v8bf*)(bp + 16);
      v16bf bf = __builtin_shufflevector(blo, bhi, 0, 1, 2, 3, 4, 5, 6, 7, 8,
                                         9, 10, 11, 12, 13, 14, 15);
      acc[g] = __builtin_amdgcn_wmma_f32_16x16x32_bf16(
          false, af, false, bf, (short)0, acc[g], false, false);
    }
  }

  // C layout: lane(0-15)=col m rows r..r+7 via vgpr r; lane(16-31)=rows +8.
  const int rb = blockIdx.x * 128 + wave * 16 + khalf * 8;
#pragma unroll
  for (int r = 0; r < 8; ++r) {
    const float dv = dinv[rb + r];
#pragma unroll
    for (int g = 0; g < G; ++g)
      Y[(size_t)(rb + r) * FCOLS + g * 16 + m] = acc[g][r] * dv;
  }
}

// ---------------------------------------------------------------------------
// h = relu(Y1 @ W1 + b1); hsT[c][k] = bf16(dinv[k] * h[k][c])  (134 MFLOP)
// One block handles a single output column c => W1 column is uniform.
// ---------------------------------------------------------------------------
__global__ __launch_bounds__(256) void k_mlp1(const float* __restrict__ Y1,
                                              const float* __restrict__ W1,
                                              const float* __restrict__ b1,
                                              const float* __restrict__ dinv,
                                              __bf16* __restrict__ hsT) {
  const int idx = blockIdx.x * 256 + threadIdx.x;  // HH * NN
  const int c = idx >> 13;
  const int k = idx & (NN - 1);
  float accv = b1[c];
#pragma unroll
  for (int j = 0; j < FIN; ++j) accv += Y1[(size_t)k * FIN + j] * W1[j * HH + c];
  const float h = fmaxf(accv, 0.0f);
  hsT[(size_t)c * NN + k] = (__bf16)(dinv[k] * h);
}

// ---------------------------------------------------------------------------
// mu/logvar heads + reparameterize; writes mu/logvar (f32 out) and z (bf16).
// ---------------------------------------------------------------------------
__global__ __launch_bounds__(256) void k_heads(
    const float* __restrict__ AH, const float* __restrict__ Wmu,
    const float* __restrict__ bmu, const float* __restrict__ Wlv,
    const float* __restrict__ blv, const float* __restrict__ eps,
    float* __restrict__ mu_out, float* __restrict__ lv_out,
    __bf16* __restrict__ zb) {
  const int idx = blockIdx.x * 256 + threadIdx.x;  // NN * ZZ
  const int k = idx >> 5;
  const int zc = idx & (ZZ - 1);
  float mu = bmu[zc], lv = blv[zc];
#pragma unroll 4
  for (int j = 0; j < HH; ++j) {
    const float a = AH[(size_t)k * HH + j];
    mu += a * Wmu[j * ZZ + zc];
    lv += a * Wlv[j * ZZ + zc];
  }
  mu_out[idx] = mu;
  lv_out[idx] = lv;
  const float z = mu + eps[idx] * __expf(0.5f * lv);
  zb[idx] = (__bf16)z;
}

// ---------------------------------------------------------------------------
// adj_recon = sigmoid(z @ z^T). K = ZZ = 32 => exactly one
// v_wmma_f32_16x16x32_bf16 per 16x16 output tile. Store-bandwidth bound.
// Block = 8 waves = 128 rows x 256 cols.
// ---------------------------------------------------------------------------
__global__ __launch_bounds__(256) void k_recon(const __bf16* __restrict__ zb,
                                               float* __restrict__ out) {
  const int lane  = threadIdx.x & 31;
  const int wave  = threadIdx.x >> 5;
  const int m     = lane & 15;
  const int khalf = lane >> 4;
  const int rowBase = blockIdx.y * 128 + wave * 16;
  const int colBase = blockIdx.x * 256;

  // A fragment: row = rowBase + m; K halves per 16-bit A layout.
  const __bf16* zr = zb + (size_t)(rowBase + m) * ZZ + khalf * 8;
  v8bf alo = *(const v8bf*)(zr);
  v8bf ahi = *(const v8bf*)(zr + 16);
  v16bf af = __builtin_shufflevector(alo, ahi, 0, 1, 2, 3, 4, 5, 6, 7, 8, 9,
                                     10, 11, 12, 13, 14, 15);
  const v8f vzero = {0.f, 0.f, 0.f, 0.f, 0.f, 0.f, 0.f, 0.f};
  const int rb = rowBase + khalf * 8;

#pragma unroll 4
  for (int t = 0; t < 16; ++t) {
    const int col = colBase + t * 16 + m;
    // B column 'col' is row 'col' of z (z z^T); transposed-load is contiguous.
    const __bf16* zc = zb + (size_t)col * ZZ + khalf * 8;
    v8bf blo = *(const v8bf*)(zc);
    v8bf bhi = *(const v8bf*)(zc + 16);
    v16bf bf = __builtin_shufflevector(blo, bhi, 0, 1, 2, 3, 4, 5, 6, 7, 8, 9,
                                       10, 11, 12, 13, 14, 15);
    v8f acc = __builtin_amdgcn_wmma_f32_16x16x32_bf16(
        false, af, false, bf, (short)0, vzero, false, false);
#pragma unroll
    for (int r = 0; r < 8; ++r) {
      const float s = 1.0f / (1.0f + __expf(-acc[r]));
      out[(size_t)(rb + r) * NN + col] = s;  // 16 lanes -> 64B contiguous
    }
  }
}

// ---------------------------------------------------------------------------
extern "C" void kernel_launch(void* const* d_in, const int* in_sizes, int n_in,
                              void* d_out, int out_size, void* d_ws,
                              size_t ws_size, hipStream_t stream) {
  (void)in_sizes; (void)n_in; (void)out_size; (void)ws_size;
  const float* x   = (const float*)d_in[0];
  const float* adj = (const float*)d_in[1];
  const float* eps = (const float*)d_in[2];
  const float* W1  = (const float*)d_in[3];
  const float* b1  = (const float*)d_in[4];
  const float* Wmu = (const float*)d_in[5];
  const float* bmu = (const float*)d_in[6];
  const float* Wlv = (const float*)d_in[7];
  const float* blv = (const float*)d_in[8];

  float* recon  = (float*)d_out;                       // [N,N]
  float* mu_out = (float*)d_out + (size_t)NN * NN;     // [N,Z]
  float* lv_out = mu_out + (size_t)NN * ZZ;            // [N,Z]

  // Workspace layout (~9.6 MB)
  char* w = (char*)d_ws;
  float*  dinv = (float*)(w);                                   // 32 KB
  __bf16* xsT  = (__bf16*)(w + (32 << 10));                     // 1 MB
  __bf16* hsT  = (__bf16*)(w + (32 << 10) + (1 << 20));         // 2 MB
  float*  Y1   = (float*) (w + (32 << 10) + (3 << 20));         // 2 MB
  float*  AH   = (float*) (w + (32 << 10) + (5 << 20));         // 4 MB
  __bf16* ZB   = (__bf16*)(w + (32 << 10) + (9 << 20));         // 512 KB

  k_deg<<<NN, 256, 0, stream>>>(adj, dinv);
  k_prep_x<<<(NN * FIN) / 256, 256, 0, stream>>>(x, dinv, xsT);
  k_gemm_big<FIN><<<NN / 128, 256, 0, stream>>>(adj, xsT, dinv, Y1);
  k_mlp1<<<(NN * HH) / 256, 256, 0, stream>>>(Y1, W1, b1, dinv, hsT);
  k_gemm_big<HH><<<NN / 128, 256, 0, stream>>>(adj, hsT, dinv, AH);
  k_heads<<<(NN * ZZ) / 256, 256, 0, stream>>>(AH, Wmu, bmu, Wlv, blv, eps,
                                               mu_out, lv_out, ZB);
  k_recon<<<dim3(NN / 256, NN / 128), 256, 0, stream>>>(ZB, recon);
}
